// epibert_68264210202944
// MI455X (gfx1250) — compile-verified, matrix operand order checked by reading, not run on hardware
//
#include <hip/hip_runtime.h>
#include <hip/hip_bf16.h>
#include <math.h>

// ---------------------------------------------------------------------------
// EpiBERT forward on gfx1250 (MI455X). All dense math routed through
// v_wmma_f32_16x16x32_bf16 (bf16 inputs converted on the fly, f32 accum).
// Register-blocked: each wave holds a 2x2 grid of 16x16 accumulators.
// ---------------------------------------------------------------------------

typedef __attribute__((ext_vector_type(16))) __bf16 v16bf;
typedef __attribute__((ext_vector_type(8)))  float  v8f;

struct U8 { unsigned int v[8]; };

__device__ __forceinline__ unsigned short f32_to_bf16(float f) {
  unsigned int x = __builtin_bit_cast(unsigned int, f);
  unsigned int r = x + 0x7FFFu + ((x >> 16) & 1u);   // round-to-nearest-even
  return (unsigned short)(r >> 16);
}

__device__ __forceinline__ float gelu_f(float x) {     // tanh approximation (jax default)
  float x3 = x * x * x;
  return 0.5f * x * (1.0f + tanhf(0.7978845608028654f * (x + 0.044715f * x3)));
}

// Load one 16x32 bf16 fragment from LDS (n/m-major, 32 bf16 per row, k-pairs
// packed so each fragment VGPR is a single b32 LDS read).
__device__ __forceinline__ v16bf load_frag(const unsigned int* s32, int rowBase, int half, int row) {
  U8 u;
#pragma unroll
  for (int i = 0; i < 8; ++i) {
    int kk = ((i < 4) ? (i << 1) : (16 + ((i - 4) << 1))) + (half << 3);
    u.v[i] = s32[(((rowBase + row) << 5) + kk) >> 1];
  }
  return __builtin_bit_cast(v16bf, u);
}

// ---------------------------------------------------------------------------
// Generic batched/strided WMMA GEMM:
//   C[b][m][n] = epi( alpha * sum_k A[b][m][k]*B[b][k][n] )
// Block: 256 threads (8 waves), tile 128(M) x 64(N); per-wave 32x32 via
// 2x2 accumulators; K stepped by 32 through LDS.
// act: 0 none, 1 relu, 2 gelu, 3 performer feature map relu(alpha*x)+1e-3.
// zdiv: optional per (batch,row) divisor.  resid: optional residual add.
// ---------------------------------------------------------------------------
__global__ void wmma_gemm(const float* __restrict__ A, int a_rs, int a_cs, long a_bs,
                          const float* __restrict__ B, int b_rs, int b_cs, long b_bs,
                          float* __restrict__ C, int c_rs, long c_bs,
                          const float* __restrict__ bias,
                          const float* __restrict__ resid,
                          const float* __restrict__ zdiv,
                          int M, int N, int K, float alpha, int act)
{
  __shared__ unsigned int As32[2048];  // 128 x 32 bf16, m-major
  __shared__ unsigned int Bs32[1024];  // 64(n) x 32(k) bf16, n-major (B staged transposed)
  unsigned short* AsU = (unsigned short*)As32;
  unsigned short* BsU = (unsigned short*)Bs32;

  const float* Ab = A + (long)blockIdx.z * a_bs;
  const float* Bb = B + (long)blockIdx.z * b_bs;

  const int tid  = threadIdx.x;
  const int wid  = tid >> 5, lane = tid & 31;
  const int wm   = wid & 3,  wn   = wid >> 2;     // 4 waves in M, 2 in N
  const int half = lane >> 4, row = lane & 15;
  const int m0 = blockIdx.x * 128, n0 = blockIdx.y * 64;

  v8f acc[2][2];
#pragma unroll
  for (int i = 0; i < 2; ++i)
#pragma unroll
    for (int j = 0; j < 2; ++j)
      acc[i][j] = (v8f){0.f, 0.f, 0.f, 0.f, 0.f, 0.f, 0.f, 0.f};

  for (int k0 = 0; k0 < K; k0 += 32) {
    // Stage A tile (zero-pad out of range -> handles K not multiple of 32).
    for (int e = tid; e < 128 * 32; e += 256) {
      int m = e >> 5, k = e & 31;
      int gm = m0 + m, gk = k0 + k;
      float v = 0.f;
      if (gm < M && gk < K) v = Ab[(long)gm * a_rs + (long)gk * a_cs];
      AsU[(m << 5) + k] = f32_to_bf16(v);
    }
    // Stage B tile transposed: Bs[n][k]; n fastest for coalescing.
    for (int e = tid; e < 64 * 32; e += 256) {
      int k = e >> 6, n = e & 63;
      int gk = k0 + k, gn = n0 + n;
      float v = 0.f;
      if (gk < K && gn < N) v = Bb[(long)gk * b_rs + (long)gn * b_cs];
      BsU[(n << 5) + k] = f32_to_bf16(v);
    }
    __syncthreads();

    v16bf af0 = load_frag(As32, (wm << 5) +  0, half, row);
    v16bf af1 = load_frag(As32, (wm << 5) + 16, half, row);
    v16bf bf0 = load_frag(Bs32, (wn << 5) +  0, half, row);
    v16bf bf1 = load_frag(Bs32, (wn << 5) + 16, half, row);

    acc[0][0] = __builtin_amdgcn_wmma_f32_16x16x32_bf16(false, af0, false, bf0, (short)0, acc[0][0], false, false);
    acc[0][1] = __builtin_amdgcn_wmma_f32_16x16x32_bf16(false, af0, false, bf1, (short)0, acc[0][1], false, false);
    acc[1][0] = __builtin_amdgcn_wmma_f32_16x16x32_bf16(false, af1, false, bf0, (short)0, acc[1][0], false, false);
    acc[1][1] = __builtin_amdgcn_wmma_f32_16x16x32_bf16(false, af1, false, bf1, (short)0, acc[1][1], false, false);
    __syncthreads();
    if (k0 + 32 < K) {
      __builtin_prefetch(Ab + (long)m0 * a_rs + (long)(k0 + 32) * a_cs, 0, 1);
      __builtin_prefetch(Bb + (long)(k0 + 32) * b_rs + (long)n0 * b_cs, 0, 1);
    }
  }

  // Epilogue. D layout: lanes 0-15 -> M=r, lanes 16-31 -> M=8+r; N = lane&15.
#pragma unroll
  for (int ni = 0; ni < 2; ++ni) {
    int gn = n0 + (wn << 5) + (ni << 4) + row;
    if (gn >= N) continue;
#pragma unroll
    for (int mi = 0; mi < 2; ++mi) {
      int gm_base = m0 + (wm << 5) + (mi << 4) + (half << 3);
#pragma unroll
      for (int r = 0; r < 8; ++r) {
        int gm = gm_base + r;
        if (gm < M) {
          float v = acc[mi][ni][r];
          if (act == 3) {
            v = fmaxf(alpha * v, 0.f) + 0.001f;         // relu(ratio*x)+STAB
          } else {
            v *= alpha;
            if (bias) v += bias[gn];
            if (act == 1) v = fmaxf(v, 0.f);
            else if (act == 2) v = gelu_f(v);
          }
          if (zdiv) v /= zdiv[(long)blockIdx.z * M + gm];
          long off = (long)blockIdx.z * c_bs + (long)gm * c_rs + gn;
          if (resid) v += resid[off];
          C[off] = v;
        }
      }
    }
  }
}

// ---------------------------------------------------------------------------
// K=5 'SAME' conv (pad 2) with WMMA. X:[L,Cin] f32, W:[5,Cin,Cout], Y:[L,Cout].
// Cin mult of 32, Cout mult of 64. Block tile 64(L) x 64(Cout); per-wave
// 16x32 via 2 accumulators. Weights staged in LDS per channel chunk.
// ---------------------------------------------------------------------------
__global__ void wmma_conv5(const float* __restrict__ X, const float* __restrict__ W,
                           const float* __restrict__ bias, float* __restrict__ Y,
                           int L, int Cin, int Cout)
{
  __shared__ unsigned int Xs32[1088];  // 68 rows x 32 ch, bf16
  __shared__ unsigned int Ws32[5120];  // 5 taps x 64(n) x 32(k), bf16
  unsigned short* XsU = (unsigned short*)Xs32;
  unsigned short* WsU = (unsigned short*)Ws32;

  const int tid  = threadIdx.x;
  const int wid  = tid >> 5, lane = tid & 31;
  const int wm   = wid & 3,  wn   = wid >> 2;
  const int half = lane >> 4, row = lane & 15;
  const int l0 = blockIdx.x * 64, n0 = blockIdx.y * 64;

  v8f acc0 = {0.f, 0.f, 0.f, 0.f, 0.f, 0.f, 0.f, 0.f};
  v8f acc1 = {0.f, 0.f, 0.f, 0.f, 0.f, 0.f, 0.f, 0.f};
  const int nchunks = Cin >> 5;

  for (int cc = 0; cc < nchunks; ++cc) {
    // Stage 68 input rows x 32 channels (pad 2 each side of 64-row tile).
    for (int e = tid; e < 68 * 32; e += 256) {
      int r = e >> 5, k = e & 31;
      int gl = l0 + r - 2;
      float v = 0.f;
      if (gl >= 0 && gl < L) v = X[(long)gl * Cin + (cc << 5) + k];
      XsU[e] = f32_to_bf16(v);
    }
    // Stage weights for all 5 taps, this chunk: Ws[tap][n][k], n fastest global.
    for (int e = tid; e < 5 * 64 * 32; e += 256) {
      int tap = e >> 11;
      int rem = e & 2047;
      int k = rem >> 6, n = rem & 63;
      float v = W[(long)(tap * Cin + (cc << 5) + k) * Cout + n0 + n];
      WsU[(((tap << 6) + n) << 5) + k] = f32_to_bf16(v);
    }
    __syncthreads();
    for (int tap = 0; tap < 5; ++tap) {
      v16bf af  = load_frag(Xs32, (wm << 4) + tap, half, row);
      v16bf bf0 = load_frag(Ws32, (tap << 6) + (wn << 5) +  0, half, row);
      v16bf bf1 = load_frag(Ws32, (tap << 6) + (wn << 5) + 16, half, row);
      acc0 = __builtin_amdgcn_wmma_f32_16x16x32_bf16(false, af, false, bf0, (short)0, acc0, false, false);
      acc1 = __builtin_amdgcn_wmma_f32_16x16x32_bf16(false, af, false, bf1, (short)0, acc1, false, false);
    }
    __syncthreads();
  }

  int gm_base = l0 + (wm << 4) + (half << 3);
#pragma unroll
  for (int ni = 0; ni < 2; ++ni) {
    int gn = n0 + (wn << 5) + (ni << 4) + row;
    if (gn >= Cout) continue;
#pragma unroll
    for (int r = 0; r < 8; ++r) {
      int gm = gm_base + r;
      if (gm < L) {
        float v = (ni == 0) ? acc0[r] : acc1[r];
        Y[(long)gm * Cout + gn] = v + bias[gn];
      }
    }
  }
}

// --------------------------- elementwise / small kernels -------------------

__global__ void bn_gelu_k(const float* __restrict__ X, const float* __restrict__ g,
                          const float* __restrict__ b, const float* __restrict__ mean,
                          const float* __restrict__ var, float* __restrict__ Y,
                          long total, int C)
{
  long idx = (long)blockIdx.x * 256 + threadIdx.x;
  if (idx >= total) return;
  int c = (int)(idx % C);
  float x = X[idx];
  float y = (x - mean[c]) * rsqrtf(var[c] + 0.001f) * g[c] + b[c];
  Y[idx] = gelu_f(y);
}

__global__ void conv_small_k(const float* __restrict__ X, const float* __restrict__ W,
                             const float* __restrict__ bias, const float* __restrict__ resid,
                             float* __restrict__ Y, int L, int Cin, int Cout,
                             int Ktap, int pad)
{
  long idx = (long)blockIdx.x * 256 + threadIdx.x;
  if (idx >= (long)L * Cout) return;
  int co = (int)(idx % Cout);
  int l  = (int)(idx / Cout);
  float acc = bias[co];
  for (int k = 0; k < Ktap; ++k) {
    int il = l + k - pad;
    if (il < 0 || il >= L) continue;
    const float* xr = X + (long)il * Cin;
    const float* wr = W + (long)k * Cin * Cout + co;
    for (int ci = 0; ci < Cin; ++ci) acc += xr[ci] * wr[(long)ci * Cout];
  }
  if (resid) acc += resid[idx];
  Y[idx] = acc;
}

__global__ void softpool_k(const float* __restrict__ X, const float* __restrict__ Lg,
                           float* __restrict__ Y, int outRows, int C)
{
  long idx = (long)blockIdx.x * 256 + threadIdx.x;
  if (idx >= (long)outRows * C) return;
  int c = (int)(idx % C);
  int l = (int)(idx / C);
  long i0 = (long)(2 * l) * C + c, i1 = i0 + C;
  float a0 = Lg[i0], a1 = Lg[i1];
  float m  = fmaxf(a0, a1);
  float w0 = expf(a0 - m), w1 = expf(a1 - m);
  Y[idx] = (X[i0] * w0 + X[i1] * w1) / (w0 + w1);
}

__global__ void maxpool_k(const float* __restrict__ X, float* __restrict__ Y,
                          int outRows, int C, int pool)
{
  long idx = (long)blockIdx.x * 256 + threadIdx.x;
  if (idx >= (long)outRows * C) return;
  int c = (int)(idx % C);
  int l = (int)(idx / C);
  float m = -3.4e38f;
  for (int p = 0; p < pool; ++p) m = fmaxf(m, X[((long)l * pool + p) * C + c]);
  Y[idx] = m;
}

__global__ void layernorm_k(const float* __restrict__ X, const float* __restrict__ g,
                            const float* __restrict__ b, float* __restrict__ Y, int C)
{
  __shared__ float red[256];
  __shared__ float s_mu, s_inv;
  int r = blockIdx.x, tid = threadIdx.x;
  const float* x = X + (long)r * C;
  float s = 0.f;
  for (int c = tid; c < C; c += 256) s += x[c];
  red[tid] = s; __syncthreads();
  for (int o = 128; o > 0; o >>= 1) { if (tid < o) red[tid] += red[tid + o]; __syncthreads(); }
  if (tid == 0) s_mu = red[0] / C;
  __syncthreads();
  float mu = s_mu;
  s = 0.f;
  for (int c = tid; c < C; c += 256) { float d = x[c] - mu; s += d * d; }
  red[tid] = s; __syncthreads();
  for (int o = 128; o > 0; o >>= 1) { if (tid < o) red[tid] += red[tid + o]; __syncthreads(); }
  if (tid == 0) s_inv = rsqrtf(red[0] / C + 1e-5f);
  __syncthreads();
  float inv = s_inv;
  float* y = Y + (long)r * C;
  for (int c = tid; c < C; c += 256) y[c] = (x[c] - mu) * inv * g[c] + b[c];
}

// In-place RoPE on [L,1152] viewed as 4 heads x 288 (144 pairs).
__global__ void rope_k(float* __restrict__ q, int L)
{
  long idx = (long)blockIdx.x * 256 + threadIdx.x;
  if (idx >= (long)L * 4 * 144) return;
  int i = (int)(idx % 144);
  int h = (int)((idx / 144) % 4);
  int l = (int)(idx / (144 * 4));
  float ang = (float)l * powf(10000.f, -(float)(2 * i) / 288.f);
  float s = sinf(ang), c = cosf(ang);
  float* p = q + (long)l * 1152 + h * 288 + 2 * i;
  float x1 = p[0], x2 = p[1];
  p[0] = x1 * c - x2 * s;
  p[1] = x1 * s + x2 * c;
}

// ks[h][m] = sum_l kp[h][l][m]   (kp: [4][512][256])
__global__ void colsum_k(const float* __restrict__ kp, float* __restrict__ ks)
{
  int idx = blockIdx.x * 256 + threadIdx.x;
  if (idx >= 4 * 256) return;
  int h = idx >> 8, m = idx & 255;
  const float* p = kp + (long)h * 512 * 256 + m;
  float s = 0.f;
  for (int l = 0; l < 512; ++l) s += p[(long)l * 256];
  ks[idx] = s;
}

// z[h][l] = dot(qp[h][l][:], ks[h][:])
__global__ void zdot_k(const float* __restrict__ qp, const float* __restrict__ ks,
                       float* __restrict__ z)
{
  int idx = blockIdx.x * 256 + threadIdx.x;
  if (idx >= 4 * 512) return;
  int h = idx / 512, l = idx % 512;
  const float* a = qp + (long)h * 512 * 256 + (long)l * 256;
  const float* b = ks + h * 256;
  float s = 0.f;
  for (int m = 0; m < 256; ++m) s += a[m] * b[m];
  z[idx] = s;
}

__global__ void motif_k(const float* __restrict__ ma, const float* __restrict__ w1,
                        const float* __restrict__ b1, const float* __restrict__ w2,
                        const float* __restrict__ b2, float* __restrict__ m8)
{
  __shared__ float h[32];
  int t = threadIdx.x;
  if (t < 32) {
    float s = b1[t];
    for (int i = 0; i < 693; ++i) s += ma[i] * w1[i * 32 + t];
    h[t] = gelu_f(s);
  }
  __syncthreads();
  if (t < 8) {
    float s = b2[t];
    for (int i = 0; i < 32; ++i) s += h[i] * w2[i * 8 + t];
    m8[t] = s;
  }
}

// hcat[512][1224] = concat(x[512][1152], a[512][64], m8[8] broadcast)
__global__ void concat_k(const float* __restrict__ x, const float* __restrict__ a,
                         const float* __restrict__ m8, float* __restrict__ hcat)
{
  int idx = blockIdx.x * 256 + threadIdx.x;
  if (idx >= 512 * 1224) return;
  int l = idx / 1224, c = idx % 1224;
  float v;
  if (c < 1152)      v = x[(long)l * 1152 + c];
  else if (c < 1216) v = a[(long)l * 64 + (c - 1152)];
  else               v = m8[c - 1216];
  hcat[idx] = v;
}

// out[r] = softplus(t2[r+2][:] . w + b), r in [0,508)
__global__ void head_k(const float* __restrict__ t2, const float* __restrict__ w,
                       const float* __restrict__ b, float* __restrict__ out)
{
  int r = blockIdx.x * 256 + threadIdx.x;
  if (r >= 508) return;
  int l = r + 2;
  float s = b[0];
  for (int i = 0; i < 192; ++i) s += t2[(long)l * 192 + i] * w[i];
  out[r] = (s > 20.f) ? s : log1pf(expf(s));
}

// ---------------------------------------------------------------------------
// Host orchestration. Input order = setup_inputs() dict insertion order,
// recursively flattened (194 tensors; indices documented inline).
// ---------------------------------------------------------------------------
extern "C" void kernel_launch(void* const* d_in, const int* in_sizes, int n_in,
                              void* d_out, int out_size, void* d_ws, size_t ws_size,
                              hipStream_t stream)
{
  (void)in_sizes; (void)n_in; (void)out_size; (void)ws_size;
#define F(i) ((const float*)d_in[(i)])
  const float* seq   = F(0);   // (1,65536,4)
  const float* atac  = F(1);   // (1,16384,1)
  const float* motif = F(2);   // (1,1,693)
  // params: 3..192 (see index math below); proj = 193.

  // ---- workspace carve (floats) ----
  float* w = (float*)d_ws;
  float* big0  = w; w += 33554432;           // 65536*512
  float* big1  = w; w += 33554432;
  float* logit = w; w += 18874368;           // 16384*1152 chunk logits
  float* a0b = w; w += 1048576;
  float* a1b = w; w += 1048576;
  float* a2b = w; w += 1048576;
  float* hbuf = w; w += 589824;              // 512*1152
  float* t0   = w; w += 589824;
  float* t1   = w; w += 589824;
  float* qb   = w; w += 589824;
  float* kb   = w; w += 589824;
  float* vb   = w; w += 589824;
  float* qpB  = w; w += 524288;              // 4*512*256
  float* kpB  = w; w += 524288;
  float* ksB  = w; w += 1024;
  float* zB   = w; w += 2048;
  float* kvB  = w; w += 294912;              // 4*256*288
  float* aoB  = w; w += 589824;
  float* f1B  = w; w += 1179648;             // 512*2304
  float* hcat = w; w += 626688;              // 512*1224
  float* t2B  = w; w += 98304;               // 512*192
  float* m8   = w; w += 8;

  auto gemm = [&](const float* A, int ars, int acs, long abs_,
                  const float* B, int brs, int bcs, long bbs,
                  float* Cp, int crs, long cbs,
                  const float* bias, const float* resid, const float* zdiv,
                  int M, int N, int K, int batch, float alpha, int act) {
    dim3 g((unsigned)((M + 127) / 128), (unsigned)((N + 63) / 64), (unsigned)batch);
    wmma_gemm<<<g, 256, 0, stream>>>(A, ars, acs, abs_, B, brs, bcs, bbs,
                                     Cp, crs, cbs, bias, resid, zdiv,
                                     M, N, K, alpha, act);
  };

  auto softmax_pool = [&](const float* x, const float* pw, float* y, int Lpre, int C) {
    const int CH = 16384;                    // chunk so logits fit in `logit`
    for (int r0 = 0; r0 < Lpre; r0 += CH) {
      int rows = (Lpre - r0 < CH) ? (Lpre - r0) : CH;
      gemm(x + (long)r0 * C, C, 1, 0, pw, C, 1, 0, logit, C, 0,
           nullptr, nullptr, nullptr, rows, C, C, 1, 1.f, 0);
      long tot = (long)(rows / 2) * C;
      softpool_k<<<(unsigned)((tot + 255) / 256), 256, 0, stream>>>(
          x + (long)r0 * C, logit, y + (long)(r0 / 2) * C, rows / 2, C);
    }
  };

  // ======================= sequence stem =======================
  {
    long tot = 65536L * 512;
    unsigned gb = (unsigned)((tot + 255) / 256);
    // 3: stem_conv_w, 4: stem_conv_b
    conv_small_k<<<gb, 256, 0, stream>>>(seq, F(3), F(4), nullptr, big0,
                                         65536, 4, 512, 15, 7);
    // 5..8: stem_res.bn {g,b,m,v}; 9: stem_res.w; 10: stem_res.b
    bn_gelu_k<<<gb, 256, 0, stream>>>(big0, F(5), F(6), F(7), F(8), big1, tot, 512);
    gemm(big1, 512, 1, 0, F(9), 512, 1, 0, big0, 512, 0,
         F(10), big0, nullptr, 65536, 512, 512, 1, 1.f, 0);    // residual add in-place
    // 11: stem_pool_w
    softmax_pool(big0, F(11), big1, 65536, 512);
  }

  // ======================= sequence tower =======================
  const int FS[6] = {512, 640, 768, 896, 1024, 1152};
  int Ls = 32768, Cin = 512;
  float* cur = big1;   // pooled stem output
  float* tmp = big0;
  for (int i = 0; i < 6; ++i) {
    int Co = FS[i];
    int base = 12 + 7 * i;  // bn.g, bn.b, bn.m, bn.v, w, b, pool_w
    long tot = (long)Ls * Cin;
    bn_gelu_k<<<(unsigned)((tot + 255) / 256), 256, 0, stream>>>(
        cur, F(base + 0), F(base + 1), F(base + 2), F(base + 3), tmp, tot, Cin);
    dim3 cg((unsigned)((Ls + 63) / 64), (unsigned)(Co / 64));
    wmma_conv5<<<cg, 256, 0, stream>>>(tmp, F(base + 4), F(base + 5), cur, Ls, Cin, Co);
    softmax_pool(cur, F(base + 6), tmp, Ls, Co);
    float* t = cur; cur = tmp; tmp = t;
    Ls >>= 1; Cin = Co;
  }
  // `cur` now holds x: [512, 1152]

  // ======================= ATAC tower =======================
  {
    long tot = 16384L * 32;
    unsigned gb = (unsigned)((tot + 255) / 256);
    conv_small_k<<<gb, 256, 0, stream>>>(atac, F(54), F(55), nullptr, a0b,
                                         16384, 1, 32, 50, 24);
    bn_gelu_k<<<gb, 256, 0, stream>>>(a0b, F(56), F(57), F(58), F(59), a1b, tot, 32);
    conv_small_k<<<gb, 256, 0, stream>>>(a1b, F(60), F(61), a0b, a0b,
                                         16384, 32, 32, 1, 0);
    long t2n = 8192L * 32;
    maxpool_k<<<(unsigned)((t2n + 255) / 256), 256, 0, stream>>>(a0b, a1b, 8192, 32, 2);
    // tower_atac[0]: 62..67 (bn g,b,m,v, w, b)
    bn_gelu_k<<<(unsigned)((t2n + 255) / 256), 256, 0, stream>>>(
        a1b, F(62), F(63), F(64), F(65), a2b, t2n, 32);
    conv_small_k<<<(unsigned)((t2n + 255) / 256), 256, 0, stream>>>(
        a2b, F(66), F(67), nullptr, a0b, 8192, 32, 32, 5, 2);
    long t4n = 2048L * 32;
    maxpool_k<<<(unsigned)((t4n + 255) / 256), 256, 0, stream>>>(a0b, a1b, 2048, 32, 4);
    // tower_atac[1]: 68..73
    bn_gelu_k<<<(unsigned)((t4n + 255) / 256), 256, 0, stream>>>(
        a1b, F(68), F(69), F(70), F(71), a2b, t4n, 32);
    long t5n = 2048L * 64;
    conv_small_k<<<(unsigned)((t5n + 255) / 256), 256, 0, stream>>>(
        a2b, F(72), F(73), nullptr, a0b, 2048, 32, 64, 5, 2);
    long t6n = 512L * 64;
    maxpool_k<<<(unsigned)((t6n + 255) / 256), 256, 0, stream>>>(a0b, a1b, 512, 64, 4);
    // a1b: [512, 64]
  }

  // motif MLP: 74 fc1_w, 75 fc1_b, 76 fc2_w, 77 fc2_b
  motif_k<<<1, 256, 0, stream>>>(motif, F(74), F(75), F(76), F(77), m8);

  // concat + pre_proj (78)
  concat_k<<<(unsigned)((512 * 1224 + 255) / 256), 256, 0, stream>>>(cur, a1b, m8, hcat);
  gemm(hcat, 1224, 1, 0, F(78), 1152, 1, 0, hbuf, 1152, 0,
       nullptr, nullptr, nullptr, 512, 1152, 1224, 1, 1.f, 0);

  // ======================= transformer (8 layers) =======================
  for (int li = 0; li < 8; ++li) {
    int B = 79 + 13 * li;  // ln1_g, ln1_b, wq, wk, wv, wo, bo, ln2_g, ln2_b,
                           // ffn_w1, ffn_b1, ffn_w2, ffn_b2
    layernorm_k<<<512, 256, 0, stream>>>(hbuf, F(B + 0), F(B + 1), t0, 1152);
    gemm(t0, 1152, 1, 0, F(B + 2), 1152, 1, 0, qb, 1152, 0,
         nullptr, nullptr, nullptr, 512, 1152, 1152, 1, 1.f, 0);
    gemm(t0, 1152, 1, 0, F(B + 3), 1152, 1, 0, kb, 1152, 0,
         nullptr, nullptr, nullptr, 512, 1152, 1152, 1, 1.f, 0);
    gemm(t0, 1152, 1, 0, F(B + 4), 1152, 1, 0, vb, 1152, 0,
         nullptr, nullptr, nullptr, 512, 1152, 1152, 1, 1.f, 0);
    long rt = 512L * 4 * 144;
    rope_k<<<(unsigned)((rt + 255) / 256), 256, 0, stream>>>(qb, 512);
    rope_k<<<(unsigned)((rt + 255) / 256), 256, 0, stream>>>(kb, 512);

    // Performer feature maps. proj = input 193, shape [256, 288]; B = proj^T.
    gemm(qb, 1152, 1, 288, F(193), 1, 288, 0, qpB, 256, 512L * 256,
         nullptr, nullptr, nullptr, 512, 256, 288, 4, 0.0625f, 3);
    gemm(kb, 1152, 1, 288, F(193), 1, 288, 0, kpB, 256, 512L * 256,
         nullptr, nullptr, nullptr, 512, 256, 288, 4, 0.0625f, 3);

    colsum_k<<<4, 256, 0, stream>>>(kpB, ksB);
    zdot_k<<<8, 256, 0, stream>>>(qpB, ksB, zB);

    // kv[h] = kp[h]^T (256x512) @ v[h] (512x288)
    gemm(kpB, 1, 256, 512L * 256, vb, 1152, 1, 288, kvB, 288, 256L * 288,
         nullptr, nullptr, nullptr, 256, 288, 512, 4, 1.f, 0);
    // out[h] = (qp[h] @ kv[h]) / z  -> written into ao at head column offset
    gemm(qpB, 256, 1, 512L * 256, kvB, 288, 1, 256L * 288, aoB, 1152, 288,
         nullptr, nullptr, zB, 512, 288, 256, 4, 1.f, 0);
    // output proj + residual
    gemm(aoB, 1152, 1, 0, F(B + 5), 1152, 1, 0, hbuf, 1152, 0,
         F(B + 6), hbuf, nullptr, 512, 1152, 1152, 1, 1.f, 0);

    // FFN
    layernorm_k<<<512, 256, 0, stream>>>(hbuf, F(B + 7), F(B + 8), t0, 1152);
    gemm(t0, 1152, 1, 0, F(B + 9), 2304, 1, 0, f1B, 2304, 0,
         F(B + 10), nullptr, nullptr, 512, 2304, 1152, 1, 1.f, 1);
    gemm(f1B, 2304, 1, 0, F(B + 11), 1152, 1, 0, hbuf, 1152, 0,
         F(B + 12), hbuf, nullptr, 512, 1152, 2304, 1, 1.f, 0);
  }

  // ======================= head =======================
  // 183 final_ln_g, 184 final_ln_b, 185..188 final_bn {g,b,m,v},
  // 189 final_conv_w, 190 final_conv_b, 191 final_dense_w, 192 final_dense_b
  layernorm_k<<<512, 256, 0, stream>>>(hbuf, F(183), F(184), t0, 1152);
  long ft = 512L * 1152;
  bn_gelu_k<<<(unsigned)((ft + 255) / 256), 256, 0, stream>>>(
      t0, F(185), F(186), F(187), F(188), t1, ft, 1152);
  gemm(t1, 1152, 1, 0, F(189), 192, 1, 0, t2B, 192, 0,
       F(190), nullptr, nullptr, 512, 192, 1152, 1, 1.f, 2);  // gelu(conv out)
  head_k<<<2, 256, 0, stream>>>(t2B, F(191), F(192), (float*)d_out);
#undef F
}
